// HierarchicalNBDL_51316269252814
// MI455X (gfx1250) — compile-verified, matrix operand order checked by reading, not run on hardware
//
#include <hip/hip_runtime.h>
#include <hip/hip_bf16.h>

// ---------------- problem constants ----------------
#define Bn   65536
#define Dn   512
#define Kn   128
#define Jn   8
#define H1n  512
#define H2n  256
#define En   32
#define FINn 544   // D + E

typedef __attribute__((ext_vector_type(16))) __bf16 v16bf;
typedef __attribute__((ext_vector_type(8)))  float  v8f;
typedef __attribute__((__vector_size__(4 * sizeof(int)))) int v4i_t;

union ABfrag {
    v16bf v;
    uint4 q[2];
};

__device__ __forceinline__ unsigned short f2bf(float f) {
    unsigned int x = __float_as_uint(f);
    unsigned int r = (x + 0x7fffu + ((x >> 16) & 1u)) >> 16;  // RNE
    return (unsigned short)r;
}

// ---- async global->LDS 16B copy (CDNA5 ASYNCcnt path), with fallbacks ----
__device__ __forceinline__ void async_cp16(const unsigned short* g, unsigned short* l) {
#if __has_builtin(__builtin_amdgcn_global_load_async_to_lds_b128)
    __builtin_amdgcn_global_load_async_to_lds_b128(
        (__attribute__((address_space(1))) v4i_t*)(g),
        (__attribute__((address_space(3))) v4i_t*)(l), 0, 0);
#elif __has_builtin(__builtin_amdgcn_load_to_lds)
    __builtin_amdgcn_load_to_lds((void*)g,
        (__attribute__((address_space(3))) void*)(l), 16, 0, 0);
#else
    *(uint4*)l = *(const uint4*)g;
#endif
}

__device__ __forceinline__ void wait_async() {
#if __has_builtin(__builtin_amdgcn_global_load_async_to_lds_b128)
# if __has_builtin(__builtin_amdgcn_s_wait_asynccnt)
    __builtin_amdgcn_s_wait_asynccnt(0);
# else
    asm volatile("s_wait_asynccnt 0x0" ::: "memory");
# endif
#endif
}

// Load a 16x32 bf16 WMMA operand fragment from LDS.
// base_row points at the start of this lane's row (A: row m; Bt: column n),
// rows are K-contiguous. Per ISA: lanes 0-15 hold K=0..7,16..23; lanes 16-31
// hold K=8..15,24..31 -> two contiguous 16B loads at +hb and +hb+16.
__device__ __forceinline__ ABfrag load_frag(const unsigned short* base_row,
                                            int koff, int lane) {
    ABfrag f;
    const int hb = (lane & 16) ? 8 : 0;
    const unsigned short* p = base_row + koff + hb;
    f.q[0] = *(const uint4*)(p);
    f.q[1] = *(const uint4*)(p + 16);
    return f;
}

__device__ __forceinline__ v8f wmma_bf16(const ABfrag& a, const ABfrag& b, v8f c) {
    return __builtin_amdgcn_wmma_f32_16x16x32_bf16(
        false, a.v, false, b.v, (short)0, c, false, false);
}

// --------- kernel 0: weight fp32 -> bf16 prep (weights TRANSPOSED) ---------
// Transposed layouts make every GEMM tile a set of contiguous 16B chunks so
// the async data mover can place them in LDS with zero shuffling.
__global__ __launch_bounds__(256) void k_prep(
    const float* __restrict__ W1, const float* __restrict__ W2,
    const float* __restrict__ Wmu, const float* __restrict__ Wlv,
    const float* __restrict__ Wz, const float* __restrict__ dicts,
    const float* __restrict__ semb,
    unsigned short* __restrict__ W1t,   // [H1][FIN]
    unsigned short* __restrict__ W2t,   // [H2][H1]
    unsigned short* __restrict__ Wht,   // [3*K][H2]  (mu|lv|z)
    unsigned short* __restrict__ Dt,    // [J][D][K]
    unsigned short* __restrict__ Sebf)  // [J][E]
{
    const int stride = gridDim.x * blockDim.x;
    const int t0 = blockIdx.x * blockDim.x + threadIdx.x;
    for (int i = t0; i < H1n * FINn; i += stride) {        // i = n*FIN + k
        int n = i / FINn, k = i % FINn;
        W1t[i] = f2bf(W1[(size_t)k * H1n + n]);
    }
    for (int i = t0; i < H2n * H1n; i += stride) {         // i = n*H1 + k
        int n = i >> 9, k = i & 511;
        W2t[i] = f2bf(W2[(size_t)k * H2n + n]);
    }
    for (int i = t0; i < 3 * Kn * H2n; i += stride) {      // i = (h*K+n)*H2 + k
        int c = i >> 8, k = i & 255;
        int h = c >> 7, n = c & 127;
        const float* src = (h == 0) ? Wmu : ((h == 1) ? Wlv : Wz);
        Wht[i] = f2bf(src[(size_t)k * Kn + n]);
    }
    for (int i = t0; i < Jn * Dn * Kn; i += stride) {      // i = (j*D+d)*K + k
        int j = i >> 16, rem = i & 65535;
        int d = rem >> 7, k = rem & 127;
        Dt[i] = f2bf(dicts[((size_t)j * Kn + k) * Dn + d]);
    }
    for (int i = t0; i < Jn * En; i += stride) Sebf[i] = f2bf(semb[i]);
}

// ---------------- kernel 1: h1 = relu([x|emb] @ W1 + b1) ----------------
__global__ __launch_bounds__(256) void k_enc1(
    const float* __restrict__ x, const int* __restrict__ sid,
    const unsigned short* __restrict__ W1t, const float* __restrict__ b1,
    const unsigned short* __restrict__ Sebf, unsigned short* __restrict__ h1)
{
    __shared__ __align__(16) unsigned short Asm[128 * 32];
    __shared__ __align__(16) unsigned short Bsm[128 * 32];  // Bt[n][k]
    const int tid = threadIdx.x, lane = tid & 31, wave = tid >> 5;
    const int wm = wave & 3, wn = wave >> 2;
    const int bm = blockIdx.y * 128, bn = blockIdx.x * 128;
    const v8f vz = {0.f,0.f,0.f,0.f,0.f,0.f,0.f,0.f};
    v8f acc[2][4];
    for (int i = 0; i < 2; i++) for (int t = 0; t < 4; t++) acc[i][t] = vz;

    for (int kb = 0; kb < 17; ++kb) {          // 544 = 16*32 (x) + 1*32 (emb)
        // B tile via async data mover: 128 n-rows x 32 bf16 from W1t
        for (int c = tid; c < 512; c += 256) {
            int n = c >> 2, part = c & 3;
            async_cp16(W1t + (size_t)(bn + n) * FINn + kb * 32 + part * 8,
                       Bsm + n * 32 + part * 8);
        }
        // A tile: fp32 x -> bf16 conversion (or site-emb gather) in VALU
        for (int idx = tid; idx < 128 * 32; idx += 256) {
            int m = idx >> 5, k = idx & 31;
            int row = bm + m;
            unsigned short v;
            if (kb < 16) v = f2bf(x[(size_t)row * Dn + kb * 32 + k]);
            else         v = Sebf[sid[row] * En + k];
            Asm[idx] = v;
        }
        if (kb < 15 && tid < 128)   // prefetch next x K-block (128B lines)
            __builtin_prefetch(x + (size_t)(bm + tid) * Dn + (kb + 1) * 32, 0, 0);
        wait_async();
        __syncthreads();
        ABfrag af[2], bfr[4];
#pragma unroll
        for (int i = 0; i < 2; i++)
            af[i] = load_frag(Asm + (wm * 32 + i * 16 + (lane & 15)) * 32, 0, lane);
#pragma unroll
        for (int t = 0; t < 4; t++)
            bfr[t] = load_frag(Bsm + (wn * 64 + t * 16 + (lane & 15)) * 32, 0, lane);
#pragma unroll
        for (int i = 0; i < 2; i++)
#pragma unroll
            for (int t = 0; t < 4; t++)
                acc[i][t] = wmma_bf16(af[i], bfr[t], acc[i][t]);
        __syncthreads();
    }
    const int hb8 = (lane & 16) ? 8 : 0;
#pragma unroll
    for (int i = 0; i < 2; i++)
        for (int t = 0; t < 4; t++)
#pragma unroll
            for (int r = 0; r < 8; r++) {
                int m = bm + wm * 32 + i * 16 + hb8 + r;
                int n = bn + wn * 64 + t * 16 + (lane & 15);
                float v = acc[i][t][r] + b1[n];
                v = v > 0.f ? v : 0.f;
                h1[(size_t)m * H1n + n] = f2bf(v);
            }
}

// ---------------- kernel 2: h2 = relu(h1 @ W2 + b2) ----------------
__global__ __launch_bounds__(256) void k_enc2(
    const unsigned short* __restrict__ h1,
    const unsigned short* __restrict__ W2t, const float* __restrict__ b2,
    unsigned short* __restrict__ h2)
{
    __shared__ __align__(16) unsigned short Asm[128 * 32];
    __shared__ __align__(16) unsigned short Bsm[128 * 32];
    const int tid = threadIdx.x, lane = tid & 31, wave = tid >> 5;
    const int wm = wave & 3, wn = wave >> 2;
    const int bm = blockIdx.y * 128, bn = blockIdx.x * 128;
    const v8f vz = {0.f,0.f,0.f,0.f,0.f,0.f,0.f,0.f};
    v8f acc[2][4];
    for (int i = 0; i < 2; i++) for (int t = 0; t < 4; t++) acc[i][t] = vz;

    for (int kb = 0; kb < 16; ++kb) {          // K = 512
        for (int c = tid; c < 512; c += 256) { // A: 128 rows x 32 bf16
            int m = c >> 2, part = c & 3;
            async_cp16(h1 + (size_t)(bm + m) * H1n + kb * 32 + part * 8,
                       Asm + m * 32 + part * 8);
        }
        for (int c = tid; c < 512; c += 256) { // B: 128 n-rows x 32 bf16
            int n = c >> 2, part = c & 3;
            async_cp16(W2t + (size_t)(bn + n) * H1n + kb * 32 + part * 8,
                       Bsm + n * 32 + part * 8);
        }
        wait_async();
        __syncthreads();
        ABfrag af[2], bfr[4];
#pragma unroll
        for (int i = 0; i < 2; i++)
            af[i] = load_frag(Asm + (wm * 32 + i * 16 + (lane & 15)) * 32, 0, lane);
#pragma unroll
        for (int t = 0; t < 4; t++)
            bfr[t] = load_frag(Bsm + (wn * 64 + t * 16 + (lane & 15)) * 32, 0, lane);
#pragma unroll
        for (int i = 0; i < 2; i++)
#pragma unroll
            for (int t = 0; t < 4; t++)
                acc[i][t] = wmma_bf16(af[i], bfr[t], acc[i][t]);
        __syncthreads();
    }
    const int hb8 = (lane & 16) ? 8 : 0;
#pragma unroll
    for (int i = 0; i < 2; i++)
        for (int t = 0; t < 4; t++)
#pragma unroll
            for (int r = 0; r < 8; r++) {
                int m = bm + wm * 32 + i * 16 + hb8 + r;
                int n = bn + wn * 64 + t * 16 + (lane & 15);
                float v = acc[i][t][r] + b2[n];
                v = v > 0.f ? v : 0.f;
                h2[(size_t)m * H2n + n] = f2bf(v);
            }
}

// ------- kernel 3: heads (mu|lv|lz) + reparam + Concrete epilogue -------
__global__ __launch_bounds__(256) void k_heads(
    const unsigned short* __restrict__ h2,
    const unsigned short* __restrict__ Wht,
    const float* __restrict__ bmu, const float* __restrict__ blv,
    const float* __restrict__ bz,
    const float* __restrict__ eps, const float* __restrict__ u,
    float* __restrict__ omu, float* __restrict__ olv, float* __restrict__ olz,
    float* __restrict__ os,  float* __restrict__ oz,
    unsigned short* __restrict__ ec)
{
    __shared__ __align__(16) unsigned short Asm[128 * 32];
    __shared__ __align__(16) unsigned short Bsm[3 * 32 * 32]; // per-head Bt[n][k]
    const int tid = threadIdx.x, lane = tid & 31, wave = tid >> 5;
    const int wm = wave & 3, wn = wave >> 2;
    const int bm = blockIdx.y * 128, bn = blockIdx.x * 32;   // latent cols
    const v8f vz = {0.f,0.f,0.f,0.f,0.f,0.f,0.f,0.f};
    v8f acc[2][3];
    for (int i = 0; i < 2; i++) for (int h = 0; h < 3; h++) acc[i][h] = vz;

    for (int kb = 0; kb < 8; ++kb) {          // K = 256
        for (int c = tid; c < 512; c += 256) { // A: 128 rows x 32 bf16
            int m = c >> 2, part = c & 3;
            async_cp16(h2 + (size_t)(bm + m) * H2n + kb * 32 + part * 8,
                       Asm + m * 32 + part * 8);
        }
        for (int c = tid; c < 384; c += 256) { // B: 3 heads x 32 n-rows x 32 bf16
            int h = c >> 7, r = (c >> 2) & 31, part = c & 3;
            async_cp16(Wht + (size_t)(h * Kn + bn + r) * H2n + kb * 32 + part * 8,
                       Bsm + h * 1024 + r * 32 + part * 8);
        }
        wait_async();
        __syncthreads();
        ABfrag af[2], bfr[3];
#pragma unroll
        for (int i = 0; i < 2; i++)
            af[i] = load_frag(Asm + (wm * 32 + i * 16 + (lane & 15)) * 32, 0, lane);
#pragma unroll
        for (int h = 0; h < 3; h++)
            bfr[h] = load_frag(Bsm + h * 1024 + (wn * 16 + (lane & 15)) * 32, 0, lane);
#pragma unroll
        for (int i = 0; i < 2; i++)
#pragma unroll
            for (int h = 0; h < 3; h++)
                acc[i][h] = wmma_bf16(af[i], bfr[h], acc[i][h]);
        __syncthreads();
    }
    const int hb8 = (lane & 16) ? 8 : 0;
    const int kl = bn + wn * 16 + (lane & 15);
#pragma unroll
    for (int i = 0; i < 2; i++)
#pragma unroll
        for (int r = 0; r < 8; r++) {
            int b = bm + wm * 32 + i * 16 + hb8 + r;
            size_t o = (size_t)b * Kn + kl;
            float mu = acc[i][0][r] + bmu[kl];
            float lv = acc[i][1][r] + blv[kl];
            float lz = acc[i][2][r] + bz[kl];
            float sv = mu + __expf(0.5f * lv) * eps[o];
            float uu = u[o];
            uu = fminf(fmaxf(uu, 1e-8f), 1.0f - 1e-8f);
            float g = __logf(uu) - log1pf(-uu);
            float zv = 1.0f / (1.0f + __expf(-2.0f * (lz + g)));  // /TEMP=0.5
            omu[o] = mu; olv[o] = lv; olz[o] = lz; os[o] = sv; oz[o] = zv;
            ec[o] = f2bf(zv * sv);
        }
}

// ------- kernel 4: x_hat = ec @ dicts[site] via row-masked accumulation -----
__global__ __launch_bounds__(256) void k_dec(
    const unsigned short* __restrict__ ec,
    const unsigned short* __restrict__ Dt, const int* __restrict__ sid,
    float* __restrict__ xhat)
{
    __shared__ __align__(16) unsigned short Asm[128 * 128]; // full K panel of ec
    __shared__ __align__(16) unsigned short Bsm[128 * 32];
    __shared__ int Ssm[128];
    const int tid = threadIdx.x, lane = tid & 31, wave = tid >> 5;
    const int wm = wave & 3, wn = wave >> 2;
    const int bm = blockIdx.y * 128, bn = blockIdx.x * 128;

    // A panel: 128 rows x 128 bf16 = 2048 x 16B chunks, async
    for (int c = tid; c < 2048; c += 256) {
        int m = c >> 4, part = c & 15;
        async_cp16(ec + (size_t)(bm + m) * Kn + part * 8,
                   Asm + m * 128 + part * 8);
    }
    if (tid < 128) Ssm[tid] = sid[bm + tid];
    wait_async();
    __syncthreads();

    const v8f vz = {0.f,0.f,0.f,0.f,0.f,0.f,0.f,0.f};
    v8f acc[2][4];
    for (int i = 0; i < 2; i++) for (int t = 0; t < 4; t++) acc[i][t] = vz;
    int mrow[2];
#pragma unroll
    for (int i = 0; i < 2; i++) mrow[i] = wm * 32 + i * 16 + (lane & 15);

    const uint4 zq = make_uint4(0u, 0u, 0u, 0u);
    for (int j = 0; j < Jn; ++j) {
        for (int kb = 0; kb < 4; ++kb) {      // K = 128
            for (int c = tid; c < 512; c += 256) { // B: 128 d-rows x 32 bf16
                int n = c >> 2, part = c & 3;
                async_cp16(Dt + ((size_t)j * Dn + bn + n) * Kn + kb * 32 + part * 8,
                           Bsm + n * 32 + part * 8);
            }
            wait_async();
            __syncthreads();
            ABfrag af[2], bfr[4];
#pragma unroll
            for (int i = 0; i < 2; i++) {
                af[i] = load_frag(Asm + mrow[i] * 128, kb * 32, lane);
                bool keep = (Ssm[mrow[i]] == j);   // zero rows not in site j
                af[i].q[0] = keep ? af[i].q[0] : zq;
                af[i].q[1] = keep ? af[i].q[1] : zq;
            }
#pragma unroll
            for (int t = 0; t < 4; t++)
                bfr[t] = load_frag(Bsm + (wn * 64 + t * 16 + (lane & 15)) * 32, 0, lane);
#pragma unroll
            for (int i = 0; i < 2; i++)
#pragma unroll
                for (int t = 0; t < 4; t++)
                    acc[i][t] = wmma_bf16(af[i], bfr[t], acc[i][t]);
            __syncthreads();
        }
    }
    const int hb8 = (lane & 16) ? 8 : 0;
#pragma unroll
    for (int i = 0; i < 2; i++)
        for (int t = 0; t < 4; t++)
#pragma unroll
            for (int r = 0; r < 8; r++) {
                int m = bm + wm * 32 + i * 16 + hb8 + r;
                int n = bn + wn * 64 + t * 16 + (lane & 15);
                xhat[(size_t)m * Dn + n] = acc[i][t][r];
            }
}

// ---------------- host launcher ----------------
extern "C" void kernel_launch(void* const* d_in, const int* in_sizes, int n_in,
                              void* d_out, int out_size, void* d_ws, size_t ws_size,
                              hipStream_t stream) {
    const float* x      = (const float*)d_in[0];
    const int*   sids   = (const int*)  d_in[1];
    const float* eps    = (const float*)d_in[2];
    const float* u      = (const float*)d_in[3];
    const float* semb   = (const float*)d_in[4];
    const float* W1     = (const float*)d_in[5];
    const float* b1     = (const float*)d_in[6];
    const float* W2     = (const float*)d_in[7];
    const float* b2     = (const float*)d_in[8];
    const float* Wmu    = (const float*)d_in[9];
    const float* bmu    = (const float*)d_in[10];
    const float* Wlv    = (const float*)d_in[11];
    const float* blv    = (const float*)d_in[12];
    const float* Wz     = (const float*)d_in[13];
    const float* bz     = (const float*)d_in[14];
    const float* dicts  = (const float*)d_in[15];

    // workspace layout (bytes, all 256B aligned)
    char* ws = (char*)d_ws;
    unsigned short* W1t  = (unsigned short*)(ws + 0);            // 512*544*2
    unsigned short* W2t  = (unsigned short*)(ws + 557056);       // 256*512*2
    unsigned short* Wht  = (unsigned short*)(ws + 819200);       // 384*256*2
    unsigned short* Dt   = (unsigned short*)(ws + 1015808);      // 8*512*128*2
    unsigned short* Sebf = (unsigned short*)(ws + 2064384);      // 8*32*2
    unsigned short* h1   = (unsigned short*)(ws + 2064896);      // B*512*2
    unsigned short* h2   = (unsigned short*)(ws + 69173760);     // B*256*2
    unsigned short* ec   = (unsigned short*)(ws + 102728192);    // B*128*2

    // output layout: x_hat, mu, logvar, logit_z, s, z (flat, fp32)
    float* out  = (float*)d_out;
    float* xhat = out;
    float* omu  = out + (size_t)Bn * Dn;
    float* olv  = omu + (size_t)Bn * Kn;
    float* olz  = olv + (size_t)Bn * Kn;
    float* os   = olz + (size_t)Bn * Kn;
    float* oz   = os  + (size_t)Bn * Kn;

    k_prep<<<1024, 256, 0, stream>>>(W1, W2, Wmu, Wlv, Wz, dicts, semb,
                                     W1t, W2t, Wht, Dt, Sebf);

    dim3 g1(H1n / 128, Bn / 128);   // (4, 512)
    k_enc1<<<g1, 256, 0, stream>>>(x, sids, W1t, b1, Sebf, h1);

    dim3 g2(H2n / 128, Bn / 128);   // (2, 512)
    k_enc2<<<g2, 256, 0, stream>>>(h1, W2t, b2, h2);

    dim3 g3(Kn / 32, Bn / 128);     // (4, 512)
    k_heads<<<g3, 256, 0, stream>>>(h2, Wht, bmu, blv, bz, eps, u,
                                    omu, olv, olz, os, oz, ec);

    dim3 g4(Dn / 128, Bn / 128);    // (4, 512)
    k_dec<<<g4, 256, 0, stream>>>(ec, Dt, sids, xhat);
}
